// CausalSelfAttentionMasked_46110768890239
// MI455X (gfx1250) — compile-verified, matrix operand order checked by reading, not run on hardware
//
#include <hip/hip_runtime.h>
#include <stdint.h>

#define B_  32
#define S_  512
#define D_  1024
#define H_  16
#define HD_ 64

typedef __attribute__((ext_vector_type(16))) __bf16 bf16x16;
typedef __attribute__((ext_vector_type(8)))  float  v8f;

union FragBF { uint32_t u[8]; bf16x16 v; uint4 q[2]; };
union FragF  { float f[8]; v8f v; };

__device__ __forceinline__ uint32_t pack2bf(float a, float b) {
  uint32_t ua = __float_as_uint(a), ub = __float_as_uint(b);
  ua += 0x7FFFu + ((ua >> 16) & 1u);          // RNE bf16
  ub += 0x7FFFu + ((ub >> 16) & 1u);
  return (ua >> 16) | (ub & 0xFFFF0000u);
}
__device__ __forceinline__ uint16_t f2bf(float a) {
  uint32_t ua = __float_as_uint(a);
  ua += 0x7FFFu + ((ua >> 16) & 1u);
  return (uint16_t)(ua >> 16);
}
__device__ __forceinline__ v8f wmma_bf16(const FragBF& a, const FragBF& b, v8f c) {
  return __builtin_amdgcn_wmma_f32_16x16x32_bf16(false, a.v, false, b.v,
                                                 (short)0, c, false, false);
}
// Flat pointer to a __shared__ object: low 32 bits == workgroup-LDS byte offset.
__device__ __forceinline__ uint32_t lds_off(const void* p) {
  return (uint32_t)(uintptr_t)p;
}
// Async 16B global -> LDS copy per lane (ASYNCcnt-tracked).
__device__ __forceinline__ void async_cp16(uint32_t lds_byte_off, const void* gaddr) {
  asm volatile("global_load_async_to_lds_b128 %0, %1, off"
               :: "v"(lds_byte_off), "v"(gaddr) : "memory");
}
__device__ __forceinline__ void wait_async0() {
  asm volatile("s_wait_asynccnt 0x0" ::: "memory");
}

// ---------------------------------------------------------------------------
// Kernel 0: one-shot f32 -> bf16 conversion (x and all weights).
// ---------------------------------------------------------------------------
__global__ __launch_bounds__(256) void cvt_bf16_kernel(
    const float* __restrict__ src, uint16_t* __restrict__ dst, int n4) {
  int i = blockIdx.x * 256 + threadIdx.x;
  if (i < n4) {
    float4 v = ((const float4*)src)[i];
    uint2 o;
    o.x = pack2bf(v.x, v.y);
    o.y = pack2bf(v.z, v.w);
    ((uint2*)dst)[i] = o;
  }
}

// ---------------------------------------------------------------------------
// Double-buffered, async-pipelined GEMM mainloop.
// C(64x64 per block) = A(bf16) @ B(bf16)^T; staged tiles are 64 rows x 64 K
// (BK=64 -> 2 WMMA K-steps per stage, 8 WMMA per wave per stage).
// Pipeline per stage: wait(own copies) -> barrier -> issue(next) -> compute.
// ---------------------------------------------------------------------------
__device__ __forceinline__ void gemm_issue(
    const uint16_t* __restrict__ Ab, const uint16_t* __restrict__ Bb,
    int m0, int n0, int k0, uint32_t aoff, uint32_t boff, int t) {
  for (int i = 0; i < 4; i++) {
    int id = i * 128 + t;               // 512 x 16B chunks per 64x64 bf16 tile
    int row = id >> 3, c = id & 7;      // 8 x 16B per row (128B row stride)
    uint32_t l = (uint32_t)(row * 128 + c * 16);
    async_cp16(aoff + l, Ab + (size_t)(m0 + row) * D_ + k0 + c * 8);
    async_cp16(boff + l, Bb + (size_t)(n0 + row) * D_ + k0 + c * 8);
  }
}

__device__ __forceinline__ void gemm_mainloop(
    const uint16_t* __restrict__ Ab, const uint16_t* __restrict__ Bb,
    int m0, int n0, uint32_t (*As)[64 * 32], uint32_t (*Bs)[64 * 32],
    FragF acc[4]) {
  const int t = threadIdx.x;
  const int wave = t >> 5, lane = t & 31;
  const int ln = lane & 15, hi = lane >> 4;
  const int khu = hi ? 4 : 0;                 // ISA 16-bit frag K layout (u32)
  uint32_t aoff[2] = { lds_off(As[0]), lds_off(As[1]) };
  uint32_t boff[2] = { lds_off(Bs[0]), lds_off(Bs[1]) };

  for (int i = 0; i < 4; i++)
    for (int r = 0; r < 8; r++) acc[i].f[r] = 0.f;

  const int NT = D_ / 64;                     // 16 staged tiles
  gemm_issue(Ab, Bb, m0, n0, 0, aoff[0], boff[0], t);

  for (int kt = 0; kt < NT; kt++) {
    wait_async0();                            // own tile-kt copies landed
    __syncthreads();                          // tile kt visible; prev readers done
    const int cur = kt & 1;
    if (kt + 1 < NT)                          // prefetch next tile during compute
      gemm_issue(Ab, Bb, m0, n0, (kt + 1) * 64, aoff[cur ^ 1], boff[cur ^ 1], t);

    for (int kk = 0; kk < 2; kk++) {          // two K=32 WMMA steps per stage
      FragBF a;
      {
        const uint32_t* ap = &As[cur][(wave * 16 + ln) * 32 + kk * 16];
        for (int i = 0; i < 4; i++) { a.u[i] = ap[khu + i]; a.u[4 + i] = ap[khu + 8 + i]; }
      }
      for (int nt = 0; nt < 4; nt++) {
        FragBF b;
        const uint32_t* bp = &Bs[cur][(nt * 16 + ln) * 32 + kk * 16];
        for (int i = 0; i < 4; i++) { b.u[i] = bp[khu + i]; b.u[4 + i] = bp[khu + 8 + i]; }
        acc[nt].v = wmma_bf16(a, b, acc[nt].v);
      }
    }
  }
}

// ---------------------------------------------------------------------------
// Kernel 1: QKV projection (y = x @ W^T) + RoPE epilogue; z selects {Q,K,V}.
// Q,K -> bf16 (B,H,S,d) roped; V -> bf16 transposed (B,H,d,S).
// ---------------------------------------------------------------------------
__global__ __launch_bounds__(128) void qkv_rope_kernel(
    const uint16_t* __restrict__ xb,
    const uint16_t* __restrict__ Wqb, const uint16_t* __restrict__ Wkb,
    const uint16_t* __restrict__ Wvb,
    uint16_t* __restrict__ Qb, uint16_t* __restrict__ Kb,
    uint16_t* __restrict__ Vt) {
  const int which = blockIdx.z;
  const uint16_t* Wb = (which == 0) ? Wqb : (which == 1) ? Wkb : Wvb;
  const int n0 = blockIdx.x * 64;
  const int m0 = blockIdx.y * 64;
  const int t = threadIdx.x;
  const int wave = t >> 5, lane = t & 31;
  const int ln = lane & 15, hi = lane >> 4;

  __shared__ __align__(16) uint32_t As[2][64 * 32];   // double-buffered 8KB tiles
  __shared__ __align__(16) uint32_t Bs[2][64 * 32];

  FragF acc[4];
  gemm_mainloop(xb, Wb, m0, n0, As, Bs, acc);

  // RoPE epilogue (reference uses tiled tables: freq index = e % 32)
  for (int nt = 0; nt < 4; nt++) {
    int n = n0 + nt * 16 + ln;
    int h = n >> 6, e = n & 63, f = e & 31;
    float invf = exp2f((float)f * (-13.287712379549449f / 32.0f)); // 10000^(-f/32)
    float sgn = (e & 1) ? 1.0f : -1.0f;
    for (int r = 0; r < 8; r++) {
      int m = m0 + wave * 16 + r + 8 * hi;
      int b = m >> 9, s = m & (S_ - 1);
      float v = acc[nt].f[r];
      if (which < 2) {
        float partner = __shfl_xor(v, 1, 32);      // pair element in lane^1
        float ang = (float)s * invf;
        float sn, c;
        __sincosf(ang, &sn, &c);
        float outv = v * c + sgn * partner * sn;
        uint16_t* dst = (which == 0) ? Qb : Kb;
        dst[(((size_t)b * H_ + h) * S_ + s) * HD_ + e] = f2bf(outv);
      } else {
        Vt[(((size_t)b * H_ + h) * HD_ + e) * S_ + s] = f2bf(v);
      }
    }
  }
}

// ---------------------------------------------------------------------------
// Kernel 2: flash attention. Block = 64 query rows of one (b,h); K/V tiles
// async-staged (double-buffered) into LDS once per block, shared by 4 waves.
// Uniform key loop; finished waves skip compute but keep staging/barriers.
// ---------------------------------------------------------------------------
__device__ __forceinline__ void attn_issue(
    const uint16_t* __restrict__ Kb, const uint16_t* __restrict__ Vt,
    int bh, int ks0, uint32_t koff, uint32_t voff, int t) {
  for (int i = 0; i < 2; i++) {
    int id = i * 128 + t;
    int kr = id >> 3, kc = id & 7;                // K tile: 32 keys x 64 e
    async_cp16(koff + (uint32_t)(kr * 128 + kc * 16),
               Kb + ((size_t)bh * S_ + ks0 + kr) * HD_ + kc * 8);
    int vr = id >> 2, vc = id & 3;                // V tile: 64 e x 32 keys
    async_cp16(voff + (uint32_t)(vr * 64 + vc * 16),
               Vt + ((size_t)bh * HD_ + vr) * S_ + ks0 + vc * 8);
  }
}

__global__ __launch_bounds__(128) void flash_attn_kernel(
    const uint16_t* __restrict__ Qb, const uint16_t* __restrict__ Kb,
    const uint16_t* __restrict__ Vt, const unsigned char* __restrict__ pm,
    uint16_t* __restrict__ Ob) {
  const int bh = blockIdx.y;
  const int b = bh >> 4, h = bh & 15;
  const int t = threadIdx.x;
  const int wave = t >> 5, lane = t & 31;
  const int ln = lane & 15, hi = lane >> 4;
  const int qblk = blockIdx.x * 64;
  const int qm0 = qblk + wave * 16;

  __shared__ __align__(16) uint32_t Ks[2][32 * 32]; // 32 keys x 64 e (bf16)
  __shared__ __align__(16) uint32_t Vs[2][64 * 16]; // 64 e x 32 keys (bf16)
  __shared__ __align__(16) uint32_t Ps[4][16 * 16]; // per-wave 16x32 P tile

  FragBF qf[2];                                     // Q 16x64 -> two K=32 frags
  {
    const uint16_t* qrow = Qb + ((size_t)bh * S_ + qm0 + ln) * HD_;
    int eh = hi ? 8 : 0;
    for (int ec = 0; ec < 2; ec++) {
      qf[ec].q[0] = *(const uint4*)(qrow + ec * 32 + eh);
      qf[ec].q[1] = *(const uint4*)(qrow + ec * 32 + eh + 16);
    }
  }

  float mrow[8], lrow[8], alpha[8];
  FragF out[4];
  for (int r = 0; r < 8; r++) { mrow[r] = -1e30f; lrow[r] = 0.f; }
  for (int tn = 0; tn < 4; tn++)
    for (int r = 0; r < 8; r++) out[tn].f[r] = 0.f;

  uint32_t koff[2] = { lds_off(Ks[0]), lds_off(Ks[1]) };
  uint32_t voff[2] = { lds_off(Vs[0]), lds_off(Vs[1]) };
  const int kend = qm0 + 16;                        // this wave's causal bound
  const int nkb = (qblk + 64) / 32;                 // block-uniform trip count

  attn_issue(Kb, Vt, bh, 0, koff[0], voff[0], t);

  for (int kb = 0; kb < nkb; kb++) {
    const int ks0 = kb * 32;
    wait_async0();
    __syncthreads();
    const int cur = kb & 1;
    if (kb + 1 < nkb)                               // prefetch next K/V tiles
      attn_issue(Kb, Vt, bh, ks0 + 32, koff[cur ^ 1], voff[cur ^ 1], t);

    if (ks0 < kend) {
      FragF sc[2];
      const int ehu = hi ? 4 : 0;
      for (int nt = 0; nt < 2; nt++) {
        for (int r = 0; r < 8; r++) sc[nt].f[r] = 0.f;
        int k = ks0 + nt * 16 + ln;
        const uint32_t* kp = &Ks[cur][(nt * 16 + ln) * 32];
        for (int ec = 0; ec < 2; ec++) {
          FragBF kf;
          const uint32_t* kq = kp + ec * 16;
          for (int i = 0; i < 4; i++) { kf.u[i] = kq[ehu + i]; kf.u[4 + i] = kq[ehu + 8 + i]; }
          sc[nt].v = wmma_bf16(qf[ec], kf, sc[nt].v);
        }
        float kmask = pm[(size_t)b * S_ + k] ? 0.f : -1e30f;
        for (int r = 0; r < 8; r++) {
          int q = qm0 + r + 8 * hi;
          float s = sc[nt].f[r] * 0.125f + kmask;   // scale = d^-0.5
          sc[nt].f[r] = (k > q) ? -1e30f : s;
        }
      }
      // online softmax; row r+8*hi lives across each 16-lane half group
      for (int r = 0; r < 8; r++) {
        float mb = fmaxf(sc[0].f[r], sc[1].f[r]);
        for (int d = 1; d < 16; d <<= 1) mb = fmaxf(mb, __shfl_xor(mb, d, 32));
        float mn = fmaxf(fmaxf(mrow[r], mb), -1e29f);
        alpha[r] = __expf(mrow[r] - mn);
        float p0 = __expf(sc[0].f[r] - mn);
        float p1 = __expf(sc[1].f[r] - mn);
        sc[0].f[r] = p0; sc[1].f[r] = p1;
        float rs = p0 + p1;
        for (int d = 1; d < 16; d <<= 1) rs += __shfl_xor(rs, d, 32);
        lrow[r] = lrow[r] * alpha[r] + rs;
        mrow[r] = mn;
      }
      for (int tn = 0; tn < 4; tn++)
        for (int r = 0; r < 8; r++) out[tn].f[r] *= alpha[r];

      // transpose P (C-layout) -> A-layout via wave-private LDS tile
      uint16_t* pw = (uint16_t*)Ps[wave];
      for (int nt = 0; nt < 2; nt++)
        for (int r = 0; r < 8; r++)
          pw[(r + 8 * hi) * 32 + nt * 16 + ln] = f2bf(sc[nt].f[r]);
      asm volatile("s_wait_dscnt 0x0" ::: "memory"); // wave-private RAW; in-order
      FragBF pa;
      {
        const uint32_t* pp = &Ps[wave][ln * 16];
        int khu = hi ? 4 : 0;
        for (int i = 0; i < 4; i++) { pa.u[i] = pp[khu + i]; pa.u[4 + i] = pp[khu + 8 + i]; }
      }
      // PV from LDS V tile
      const int ksh = hi ? 4 : 0;
      for (int tn = 0; tn < 4; tn++) {
        FragBF vf;
        const uint32_t* vp = &Vs[cur][(tn * 16 + ln) * 16];
        for (int i = 0; i < 4; i++) { vf.u[i] = vp[ksh + i]; vf.u[4 + i] = vp[ksh + 8 + i]; }
        out[tn].v = wmma_bf16(pa, vf, out[tn].v);
      }
    }
  }

  for (int r = 0; r < 8; r++) {
    float inv = lrow[r] > 0.f ? 1.0f / lrow[r] : 0.f;
    int s = qm0 + r + 8 * hi;
    uint16_t* orow = Ob + ((size_t)b * S_ + s) * D_ + h * HD_;
    for (int tn = 0; tn < 4; tn++)
      orow[tn * 16 + ln] = f2bf(out[tn].f[r] * inv);
  }
}

// ---------------------------------------------------------------------------
// Kernel 3: output projection Y = attnOut @ Wo^T + bo (f32 out).
// ---------------------------------------------------------------------------
__global__ __launch_bounds__(128) void out_proj_kernel(
    const uint16_t* __restrict__ A, const uint16_t* __restrict__ Wob,
    const float* __restrict__ bo, float* __restrict__ Y) {
  const int n0 = blockIdx.x * 64;
  const int m0 = blockIdx.y * 64;
  const int t = threadIdx.x;
  const int wave = t >> 5, lane = t & 31;
  const int ln = lane & 15, hi = lane >> 4;

  __shared__ __align__(16) uint32_t As[2][64 * 32];
  __shared__ __align__(16) uint32_t Bs[2][64 * 32];

  FragF acc[4];
  gemm_mainloop(A, Wob, m0, n0, As, Bs, acc);

  for (int nt = 0; nt < 4; nt++) {
    int n = n0 + nt * 16 + ln;
    float bias = bo[n];
    for (int r = 0; r < 8; r++) {
      int m = m0 + wave * 16 + r + 8 * hi;
      Y[(size_t)m * D_ + n] = acc[nt].f[r] + bias;
    }
  }
}

extern "C" void kernel_launch(void* const* d_in, const int* in_sizes, int n_in,
                              void* d_out, int out_size, void* d_ws, size_t ws_size,
                              hipStream_t stream) {
  const float* x  = (const float*)d_in[0];
  const unsigned char* pm = (const unsigned char*)d_in[1];  // bool mask
  const float* Wq = (const float*)d_in[2];
  const float* Wk = (const float*)d_in[3];
  const float* Wv = (const float*)d_in[4];
  const float* Wo = (const float*)d_in[5];
  const float* bo = (const float*)d_in[6];
  float* Y = (float*)d_out;

  const size_t elems = (size_t)B_ * H_ * S_ * HD_;  // 16M (== B*S*D)
  const size_t welem = (size_t)D_ * D_;             // 1M per weight
  uint16_t* Qb  = (uint16_t*)d_ws;                  // bf16 (B,H,S,d), roped
  uint16_t* Kb  = Qb + elems;                       // bf16 (B,H,S,d), roped
  uint16_t* Vt  = Kb + elems;                       // bf16 (B,H,d,S)
  uint16_t* Ob  = Vt + elems;                       // bf16 (B,S,D) attn out
  uint16_t* xb  = Ob + elems;                       // bf16 copy of x
  uint16_t* Wqb = xb + elems;
  uint16_t* Wkb = Wqb + welem;
  uint16_t* Wvb = Wkb + welem;
  uint16_t* Wob = Wvb + welem;

  // one-shot bf16 conversion of all GEMM operands
  cvt_bf16_kernel<<<(int)(elems / 4 + 255) / 256, 256, 0, stream>>>(x, xb, (int)(elems / 4));
  cvt_bf16_kernel<<<(int)(welem / 4 + 255) / 256, 256, 0, stream>>>(Wq, Wqb, (int)(welem / 4));
  cvt_bf16_kernel<<<(int)(welem / 4 + 255) / 256, 256, 0, stream>>>(Wk, Wkb, (int)(welem / 4));
  cvt_bf16_kernel<<<(int)(welem / 4 + 255) / 256, 256, 0, stream>>>(Wv, Wvb, (int)(welem / 4));
  cvt_bf16_kernel<<<(int)(welem / 4 + 255) / 256, 256, 0, stream>>>(Wo, Wob, (int)(welem / 4));

  qkv_rope_kernel<<<dim3(D_ / 64, (B_ * S_) / 64, 3), 128, 0, stream>>>(
      xb, Wqb, Wkb, Wvb, Qb, Kb, Vt);
  flash_attn_kernel<<<dim3(S_ / 64, B_ * H_), 128, 0, stream>>>(
      Qb, Kb, Vt, pm, Ob);
  out_proj_kernel<<<dim3(D_ / 64, (B_ * S_) / 64), 128, 0, stream>>>(
      Ob, Wob, bo, Y);
}
